// BiLSTM_CRF_81561428951666
// MI455X (gfx1250) — matrix operand order repaired and checked by reading, not compiled, not run on hardware
//
#include <hip/hip_runtime.h>
#include <hip/hip_bf16.h>

typedef __attribute__((ext_vector_type(16))) _Float16 v16h;
typedef __attribute__((ext_vector_type(8)))  float    v8f;
typedef __attribute__((ext_vector_type(4)))  unsigned u32x4;
typedef __attribute__((ext_vector_type(8)))  int      i32x8;
typedef __attribute__((ext_vector_type(4)))  int      i32x4;

#define T_LEN   4096
#define EMB     300
#define EMBP    320     // K padded to multiple of 32
#define HID     512
#define G4      2048    // 4*HID
#define NOUT    4096    // 4*HID * 2 directions
#define KTAG    20
#define KP      32      // tag dim padded for WMMA N tiles
#define NEGV    (-10000.0f)
#define START_T 18
#define STOP_T  19

// ---------------------------------------------------------------------------
// Pack: gather embedding rows + convert GEMM operands to f16, zero-padded.
// ---------------------------------------------------------------------------
__global__ void pack_kernel(const int* __restrict__ sentence,
                            const float* __restrict__ embed,
                            const float* __restrict__ w_ih_f,
                            const float* __restrict__ w_ih_b,
                            const float* __restrict__ W_tag,
                            _Float16* __restrict__ Xh,
                            _Float16* __restrict__ Wh,
                            _Float16* __restrict__ WtagH) {
    int row = blockIdx.x;
    int tid = threadIdx.x;
    if (row < T_LEN) {
        int src = sentence[row];
        const float* s = embed + (size_t)src * EMB;
        _Float16* d = Xh + (size_t)row * EMBP;
        for (int e = tid; e < EMBP; e += blockDim.x)
            d[e] = (_Float16)(e < EMB ? s[e] : 0.0f);
    } else if (row < T_LEN + NOUT) {
        int r = row - T_LEN;
        const float* s = (r < G4) ? (w_ih_f + (size_t)r * EMB)
                                  : (w_ih_b + (size_t)(r - G4) * EMB);
        _Float16* d = Wh + (size_t)r * EMBP;
        for (int e = tid; e < EMBP; e += blockDim.x)
            d[e] = (_Float16)(e < EMB ? s[e] : 0.0f);
    } else {
        int r = row - T_LEN - NOUT;
        _Float16* d = WtagH + (size_t)r * (2 * HID);
        for (int e = tid; e < 2 * HID; e += blockDim.x)
            d[e] = (_Float16)(r < KTAG ? W_tag[(size_t)r * (2 * HID) + e] : 0.0f);
    }
}

// ---------------------------------------------------------------------------
// WMMA fragment index helpers (CDNA5 ISA 7.12.2, wave32):
//   A 16x32 f16 : lane m=l%16, vgpr i holds k = 16*(i/4) + 8*(l/16) + 2*(i%4)
//   B 32x16 f16 : lane n=l%16, vgpr i holds k = 16*(i/4) + 8*(l/16) + 2*(i%4)
//   D f32       : vgpr r holds (m = r + 8*(l/16), n = l%16)
// ---------------------------------------------------------------------------
__device__ __forceinline__ void load_a_frag(const _Float16* __restrict__ arow,
                                            int kc, int hf, unsigned* u) {
#pragma unroll
    for (int i = 0; i < 8; ++i) {
        int ka = kc + 16 * (i >> 2) + 8 * hf + 2 * (i & 3);
        u[i] = *(const unsigned*)(arow + ka);
    }
}
__device__ __forceinline__ void load_b_frag(const _Float16* __restrict__ brow,
                                            int kc, int hf, unsigned* u) {
#pragma unroll
    for (int i = 0; i < 8; ++i) {
        int kb = kc + 16 * (i >> 2) + 8 * hf + 2 * (i & 3);
        u[i] = *(const unsigned*)(brow + kb);
    }
}

__device__ __forceinline__ v8f wmma_tile16(const _Float16* __restrict__ Abase,
                                           const _Float16* __restrict__ Bbase,
                                           int lda, int ldb, int kd, int lane) {
    const int hf = (lane >> 4) & 1;
    const int mr = lane & 15;
    const _Float16* arow = Abase + (size_t)mr * lda;
    const _Float16* brow = Bbase + (size_t)mr * ldb;
    v8f acc = {};
    for (int kc = 0; kc < kd; kc += 32) {
        union { v16h v; unsigned u[8]; } a, b;
        load_a_frag(arow, kc, hf, a.u);
        load_b_frag(brow, kc, hf, b.u);
        acc = __builtin_amdgcn_wmma_f32_16x16x32_f16(false, a.v, false, b.v,
                                                     (short)0, acc, false, false);
    }
    return acc;
}

// ---------------------------------------------------------------------------
// x_proj[T, NOUT] = Xh @ Wh^T + (b_ih + b_hh).
// Block (128 thr, 4 waves) owns one M-tile (16 rows of Xh) staged in LDS via
// the Tensor Data Mover (fallback: cooperative copy); each wave computes 4
// N-tiles, reusing one A fragment across 4 WMMAs per k-chunk.
// ---------------------------------------------------------------------------
__global__ void xproj_gemm(const _Float16* __restrict__ Xh,
                           const _Float16* __restrict__ Wh,
                           const float* __restrict__ b_ih_f,
                           const float* __restrict__ b_hh_f,
                           const float* __restrict__ b_ih_b,
                           const float* __restrict__ b_hh_b,
                           float* __restrict__ xproj) {
    __shared__ __align__(16) _Float16 hsA[16 * EMBP];   // 10 KB
    const int wave = threadIdx.x >> 5;
    const int lane = threadIdx.x & 31;
    const int tm = blockIdx.x >> 4;        // 256 M-tiles
    const int ng = blockIdx.x & 15;        // 16 N-groups of 256 cols
    const int tn0 = ng * 16 + wave * 4;    // 4 N-tiles per wave

#if __has_builtin(__builtin_amdgcn_tensor_load_to_lds)
    if (threadIdx.x < 32) {
        // --- TDM descriptor: 1D tile of 2560 dwords, global -> LDS ---
        unsigned lds_off = (unsigned)(size_t)(&hsA[0]);
        unsigned long long ga =
            (unsigned long long)(size_t)(Xh + (size_t)(tm * 16) * EMBP);
        u32x4 g0;
        g0.x = 1u;                                    // count=1 (valid, user)
        g0.y = lds_off;                               // lds_addr (bytes)
        g0.z = (unsigned)(ga & 0xFFFFFFFFu);          // global_addr[31:0]
        g0.w = (unsigned)((ga >> 32) & 0x01FFFFFFu)   // global_addr[56:32]
             | (2u << 30);                            // type=2 (image)
        i32x8 g1;
        g1[0] = 0x00020000;                 // data_size=2 (4B), no flags
        g1[1] = (int)((2560u & 0xFFFFu) << 16);   // tensor_dim0 lo16
        g1[2] = (int)(1u << 16);            // tensor_dim0 hi=0 | tensor_dim1=1 lo
        g1[3] = (int)(2560u << 16);         // tensor_dim1 hi=0 | tile_dim0=2560
        g1[4] = 1;                          // tile_dim1=1, tile_dim2=0
        g1[5] = 2560;                       // tensor_dim0_stride lo32
        g1[6] = (int)((2560u & 0xFFFFu) << 16); // stride0 hi=0 | stride1 lo16
        g1[7] = 0;                          // stride1 hi
        i32x4 z4 = {0, 0, 0, 0};
#if defined(__clang_major__) && (__clang_major__ >= 23)
        i32x8 z8 = {0, 0, 0, 0, 0, 0, 0, 0};
        __builtin_amdgcn_tensor_load_to_lds(g0, g1, z4, z4, z8, 0);
#else
        __builtin_amdgcn_tensor_load_to_lds(g0, g1, z4, z4, 0);
#endif
#if __has_builtin(__builtin_amdgcn_s_wait_tensorcnt)
        __builtin_amdgcn_s_wait_tensorcnt((short)0);
#else
        asm volatile("s_wait_tensorcnt 0x0" ::: "memory");
#endif
    }
#else
    {   // fallback: cooperative 16B copies
        const uint4* src = (const uint4*)(Xh + (size_t)(tm * 16) * EMBP);
        uint4* dst = (uint4*)hsA;
        for (int i = threadIdx.x; i < 16 * EMBP / 8; i += blockDim.x)
            dst[i] = src[i];
    }
#endif
    __syncthreads();

    const int hf = lane >> 4;
    const int mr = lane & 15;
    const _Float16* arow = hsA + (size_t)mr * EMBP;
    const _Float16* brow0 = Wh + (size_t)(tn0 * 16 + mr) * EMBP;

    v8f acc0 = {}, acc1 = {}, acc2 = {}, acc3 = {};
    for (int kc = 0; kc < EMBP; kc += 32) {
        union { v16h v; unsigned u[8]; } a, b0, b1, b2, b3;
        load_a_frag(arow, kc, hf, a.u);                 // from LDS
        load_b_frag(brow0,                kc, hf, b0.u);
        load_b_frag(brow0 + 16 * EMBP,    kc, hf, b1.u);
        load_b_frag(brow0 + 32 * EMBP,    kc, hf, b2.u);
        load_b_frag(brow0 + 48 * EMBP,    kc, hf, b3.u);
        acc0 = __builtin_amdgcn_wmma_f32_16x16x32_f16(false, a.v, false, b0.v,
                                                      (short)0, acc0, false, false);
        acc1 = __builtin_amdgcn_wmma_f32_16x16x32_f16(false, a.v, false, b1.v,
                                                      (short)0, acc1, false, false);
        acc2 = __builtin_amdgcn_wmma_f32_16x16x32_f16(false, a.v, false, b2.v,
                                                      (short)0, acc2, false, false);
        acc3 = __builtin_amdgcn_wmma_f32_16x16x32_f16(false, a.v, false, b3.v,
                                                      (short)0, acc3, false, false);
    }

    v8f accs[4] = {acc0, acc1, acc2, acc3};
#pragma unroll
    for (int p = 0; p < 4; ++p) {
        int n = (tn0 + p) * 16 + mr;
        float bias = (n < G4) ? (b_ih_f[n] + b_hh_f[n])
                              : (b_ih_b[n - G4] + b_hh_b[n - G4]);
#pragma unroll
        for (int r = 0; r < 8; ++r) {
            int m = tm * 16 + r + 8 * hf;
            xproj[(size_t)m * NOUT + n] = accs[p][r] + bias;
        }
    }
}

// ---------------------------------------------------------------------------
// Sequential LSTM: blockIdx.x = direction, 512 threads; thread j owns cell j
// (4 rows of w_hh matvec, L2-resident) against h in LDS.
// ---------------------------------------------------------------------------
__global__ void lstm_kernel(const float* __restrict__ xproj,
                            const float* __restrict__ w_hh_f,
                            const float* __restrict__ w_hh_b,
                            const float* __restrict__ h0,
                            const float* __restrict__ c0,
                            _Float16* __restrict__ hidden) {
    __shared__ __align__(16) float hs[HID];
    const int dir = blockIdx.x;
    const int j = threadIdx.x;
    const float* whh = dir ? w_hh_b : w_hh_f;
    float c = c0[dir * HID + j];
    hs[j] = h0[dir * HID + j];
    __syncthreads();
    const float4* w0 = (const float4*)(whh + (size_t)j * HID);
    const float4* w1 = (const float4*)(whh + (size_t)(HID + j) * HID);
    const float4* w2 = (const float4*)(whh + (size_t)(2 * HID + j) * HID);
    const float4* w3 = (const float4*)(whh + (size_t)(3 * HID + j) * HID);
    for (int s = 0; s < T_LEN; ++s) {
        const int t = dir ? (T_LEN - 1 - s) : s;
        const float* xp = xproj + (size_t)t * NOUT + dir * G4;
        float a0 = xp[j], a1 = xp[HID + j], a2 = xp[2 * HID + j], a3 = xp[3 * HID + j];
        const int tnx = dir ? (t - 1) : (t + 1);
        if (tnx >= 0 && tnx < T_LEN)
            __builtin_prefetch(xproj + (size_t)tnx * NOUT + dir * G4 + j, 0, 1);
        const float4* hv4 = (const float4*)hs;
        for (int kk = 0; kk < HID / 4; ++kk) {
            float4 h4 = hv4[kk];
            float4 x0 = w0[kk], x1 = w1[kk], x2 = w2[kk], x3 = w3[kk];
            a0 += x0.x * h4.x + x0.y * h4.y + x0.z * h4.z + x0.w * h4.w;
            a1 += x1.x * h4.x + x1.y * h4.y + x1.z * h4.z + x1.w * h4.w;
            a2 += x2.x * h4.x + x2.y * h4.y + x2.z * h4.z + x2.w * h4.w;
            a3 += x3.x * h4.x + x3.y * h4.y + x3.z * h4.z + x3.w * h4.w;
        }
        float ig = 1.0f / (1.0f + __expf(-a0));
        float fg = 1.0f / (1.0f + __expf(-a1));
        float gg = tanhf(a2);
        float og = 1.0f / (1.0f + __expf(-a3));
        c = fg * c + ig * gg;
        float h = og * tanhf(c);
        __syncthreads();
        hs[j] = h;
        hidden[(size_t)t * (2 * HID) + dir * HID + j] = (_Float16)h;
        __syncthreads();
    }
}

// ---------------------------------------------------------------------------
// feats[T, KP] = hidden[T,1024] @ WtagH[KP,1024]^T + b_tag
// ---------------------------------------------------------------------------
__global__ void feats_gemm(const _Float16* __restrict__ hidden,
                           const _Float16* __restrict__ WtagH,
                           const float* __restrict__ b_tag,
                           float* __restrict__ feats) {
    const int wave = threadIdx.x >> 5;
    const int lane = threadIdx.x & 31;
    const int tile = blockIdx.x * 4 + wave;
    const int tnc = KP / 16;  // 2
    const int tm = tile / tnc, tn = tile % tnc;
    v8f acc = wmma_tile16(hidden + (size_t)(tm * 16) * (2 * HID),
                          WtagH + (size_t)(tn * 16) * (2 * HID),
                          2 * HID, 2 * HID, 2 * HID, lane);
    const int hf = lane >> 4;
    const int n = tn * 16 + (lane & 15);
    float bias = (n < KTAG) ? b_tag[n] : 0.0f;
#pragma unroll
    for (int r = 0; r < 8; ++r) {
        int m = tm * 16 + r + 8 * hf;
        feats[(size_t)m * KP + n] = acc[r] + bias;
    }
}

// ---------------------------------------------------------------------------
// CRF forward (logsumexp over T) + gold path score. One wave32.
// ---------------------------------------------------------------------------
__global__ void crf_kernel(const float* __restrict__ feats,
                           const float* __restrict__ trans,
                           const int* __restrict__ tags,
                           float* __restrict__ out) {
    __shared__ float al[KTAG];
    const int lane = threadIdx.x;
    float tr[KTAG];
#pragma unroll
    for (int i = 0; i < KTAG; ++i) tr[i] = 0.0f;
    if (lane < KTAG) {
#pragma unroll
        for (int i = 0; i < KTAG; ++i) tr[i] = trans[lane * KTAG + i];
        al[lane] = (lane == START_T) ? 0.0f : NEGV;
    }
    __syncthreads();
    for (int t = 0; t < T_LEN; ++t) {
        float na = 0.0f;
        if (lane < KTAG) {
            float v[KTAG];
            float m = -3.4e38f;
#pragma unroll
            for (int i = 0; i < KTAG; ++i) { v[i] = al[i] + tr[i]; m = fmaxf(m, v[i]); }
            float s = 0.0f;
#pragma unroll
            for (int i = 0; i < KTAG; ++i) s += __expf(v[i] - m);
            na = m + __logf(s) + feats[(size_t)t * KP + lane];
        }
        __syncthreads();
        if (lane < KTAG) al[lane] = na;
        __syncthreads();
    }
    float v = (lane < KTAG) ? (al[lane] + trans[STOP_T * KTAG + lane]) : NEGV;
    float m = v;
    for (int o = 16; o > 0; o >>= 1) m = fmaxf(m, __shfl_xor(m, o, 32));
    float e = __expf(v - m);
    for (int o = 16; o > 0; o >>= 1) e += __shfl_xor(e, o, 32);
    float fwd = m + __logf(e);
    float g = 0.0f;
    for (int t = lane; t < T_LEN; t += 32) {
        int cur = tags[t];
        int prev = (t == 0) ? START_T : tags[t - 1];
        g += trans[cur * KTAG + prev] + feats[(size_t)t * KP + cur];
    }
    for (int o = 16; o > 0; o >>= 1) g += __shfl_xor(g, o, 32);
    if (lane == 0) {
        g += trans[STOP_T * KTAG + tags[T_LEN - 1]];
        out[0] = fwd - g;
    }
}

// ---------------------------------------------------------------------------
extern "C" void kernel_launch(void* const* d_in, const int* in_sizes, int n_in,
                              void* d_out, int out_size, void* d_ws, size_t ws_size,
                              hipStream_t stream) {
    const int*   sentence = (const int*)d_in[0];
    const int*   tags     = (const int*)d_in[1];
    const float* embed    = (const float*)d_in[2];
    const float* w_ih_f   = (const float*)d_in[3];
    const float* w_hh_f   = (const float*)d_in[4];
    const float* b_ih_f   = (const float*)d_in[5];
    const float* b_hh_f   = (const float*)d_in[6];
    const float* w_ih_b   = (const float*)d_in[7];
    const float* w_hh_b   = (const float*)d_in[8];
    const float* b_ih_b   = (const float*)d_in[9];
    const float* b_hh_b   = (const float*)d_in[10];
    const float* h0       = (const float*)d_in[11];
    const float* c0       = (const float*)d_in[12];
    const float* W_tag    = (const float*)d_in[13];
    const float* b_tag    = (const float*)d_in[14];
    const float* trans    = (const float*)d_in[15];

    _Float16* Xh     = (_Float16*)d_ws;                          // T*EMBP f16
    _Float16* Wh     = Xh + (size_t)T_LEN * EMBP;                // NOUT*EMBP f16
    float*    xproj  = (float*)(Wh + (size_t)NOUT * EMBP);       // T*NOUT f32
    _Float16* hidden = (_Float16*)(xproj + (size_t)T_LEN * NOUT);// T*1024 f16
    _Float16* WtagH  = hidden + (size_t)T_LEN * 2 * HID;         // KP*1024 f16
    float*    feats  = (float*)(WtagH + (size_t)KP * 2 * HID);   // T*KP f32

    pack_kernel<<<T_LEN + NOUT + KP, 64, 0, stream>>>(
        sentence, embed, w_ih_f, w_ih_b, W_tag, Xh, Wh, WtagH);

    // 256 M-tiles x 16 N-groups; each block: 1 M-tile x 256 cols
    xproj_gemm<<<256 * 16, 128, 0, stream>>>(
        Xh, Wh, b_ih_f, b_hh_f, b_ih_b, b_hh_b, xproj);

    lstm_kernel<<<2, HID, 0, stream>>>(xproj, w_hh_f, w_hh_b, h0, c0, hidden);

    feats_gemm<<<512 / 4, 128, 0, stream>>>(hidden, WtagH, b_tag, feats);

    crf_kernel<<<1, 32, 0, stream>>>(feats, trans, tags, (float*)d_out);
}
